// FMoE_77292231459432
// MI455X (gfx1250) — matrix operand order repaired and checked by reading, not compiled, not run on hardware
//
#include <hip/hip_runtime.h>

#define NUM_EXPERT 8
#define IN_FEAT    1024
#define OUT_FEAT   1024
#define N_TOKENS   8192
#define SEG_ALIGN  64
#define MAX_ROWS   (N_TOKENS + NUM_EXPERT * SEG_ALIGN)   // 8704 worst-case padded rows
#define MAX_MTILES (MAX_ROWS / 64)                       // 136
#define K_STEP     32
#define BN         128                                   // block N tile
#define LDS_ROW    40                                    // 32 K-elems + 8 pad (80B row, debanks LDS)

typedef __attribute__((ext_vector_type(16))) __bf16 bf16x16;
typedef __attribute__((ext_vector_type(8)))  __bf16 bf16x8;
typedef __attribute__((ext_vector_type(8)))  float  f32x8;
typedef __attribute__((ext_vector_type(4)))  float  f32x4;
typedef int v4i_t __attribute__((vector_size(16)));      // matches async-builtin param type
#define AS1 __attribute__((address_space(1)))
#define AS3 __attribute__((address_space(3)))

// ---- workspace layout ----
// meta ints: [0..7]=counts, [8..15]=fill, [16..24]=segStart, [32..]=perm[MAX_ROWS]
#define OFF_ABF   ((size_t)64 * 1024)
#define ABF_BYTES ((size_t)MAX_ROWS * IN_FEAT * 2)             // ~17.8 MB
#define OFF_WBF   (OFF_ABF + ABF_BYTES)
#define WBF_BYTES ((size_t)NUM_EXPERT * OUT_FEAT * IN_FEAT * 2) // 16 MB
#define OFF_DUMP  (OFF_WBF + WBF_BYTES)                         // 1 KB dump line for pad-row stores

// ---- CDNA5 async global->LDS path (compile-time feature test, safe fallback) ----
#if defined(__HIP_DEVICE_COMPILE__) && __has_builtin(__builtin_amdgcn_global_load_async_to_lds_b128)
#define HAVE_ASYNC 1
#else
#define HAVE_ASYNC 0
#endif

#if HAVE_ASYNC
#if __has_builtin(__builtin_amdgcn_s_wait_asynccnt)
#define WAIT_ASYNC() __builtin_amdgcn_s_wait_asynccnt(0)
#else
#define WAIT_ASYNC() asm volatile("s_wait_asynccnt 0x0" ::: "memory")
#endif
#else
#define WAIT_ASYNC() ((void)0)
#endif

__global__ void k_init(int* meta) {
    if (threadIdx.x < 16) meta[threadIdx.x] = 0;   // counts + fill
}

__global__ void k_count(const int* __restrict__ gate, int* __restrict__ counts) {
    int i = blockIdx.x * blockDim.x + threadIdx.x;
    if (i < N_TOKENS) atomicAdd(&counts[gate[i]], 1);
}

__global__ void k_scan(const int* __restrict__ counts, int* __restrict__ segStart) {
    if (threadIdx.x == 0) {
        int acc = 0;
        for (int e = 0; e < NUM_EXPERT; ++e) {
            segStart[e] = acc;
            acc += (counts[e] + SEG_ALIGN - 1) & ~(SEG_ALIGN - 1);
        }
        segStart[NUM_EXPERT] = acc;
    }
}

// fp32 weight -> bf16 once per launch (stays resident in the 192MB L2 for the GEMM)
__global__ void k_wconv(const float* __restrict__ w, __bf16* __restrict__ wbf) {
    size_t i = ((size_t)blockIdx.x * blockDim.x + threadIdx.x) * 8;
    f32x4 f0 = *(const f32x4*)(w + i);
    f32x4 f1 = *(const f32x4*)(w + i + 4);
    bf16x8 o;
    o[0] = (__bf16)f0[0]; o[1] = (__bf16)f0[1]; o[2] = (__bf16)f0[2]; o[3] = (__bf16)f0[3];
    o[4] = (__bf16)f1[0]; o[5] = (__bf16)f1[1]; o[6] = (__bf16)f1[2]; o[7] = (__bf16)f1[3];
    *(bf16x8*)(wbf + i) = o;
}

// one block per token: claim packed slot in its expert bucket, convert row fp32->bf16
__global__ void k_gather(const float* __restrict__ inp, const int* __restrict__ gate,
                         const int* __restrict__ segStart, int* __restrict__ fill,
                         int* __restrict__ perm, __bf16* __restrict__ Abf) {
    __shared__ int s_pos;
    const int tok = blockIdx.x;
    if (threadIdx.x == 0) {
        int e   = gate[tok];
        int pos = segStart[e] + atomicAdd(&fill[e], 1);
        perm[pos] = tok;
        s_pos = pos;
    }
    __syncthreads();
    const int pos = s_pos;
    const float* src = inp + (size_t)tok * IN_FEAT + threadIdx.x * 8;
    __bf16*      dst = Abf + (size_t)pos * IN_FEAT + threadIdx.x * 8;
    f32x4 f0 = *(const f32x4*)(src);
    f32x4 f1 = *(const f32x4*)(src + 4);
    bf16x8 o;
    o[0] = (__bf16)f0[0]; o[1] = (__bf16)f0[1]; o[2] = (__bf16)f0[2]; o[3] = (__bf16)f0[3];
    o[4] = (__bf16)f1[0]; o[5] = (__bf16)f1[1]; o[6] = (__bf16)f1[2]; o[7] = (__bf16)f1[3];
    *(bf16x8*)dst = o;
}

// one block per padded row: zero pad rows so WMMA on them is harmless
__global__ void k_padzero(const int* __restrict__ segStart, const int* __restrict__ counts,
                          int* __restrict__ perm, __bf16* __restrict__ Abf) {
    const int row = blockIdx.x;
    if (row >= segStart[NUM_EXPERT]) return;
    int e = 0;
#pragma unroll
    for (int i = 1; i < NUM_EXPERT; ++i) if (row >= segStart[i]) e = i;
    if (row - segStart[e] < counts[e]) return;
    if (threadIdx.x == 0) perm[row] = -1;
    bf16x8 z;
#pragma unroll
    for (int i = 0; i < 8; ++i) z[i] = (__bf16)0.0f;
    *(bf16x8*)(Abf + (size_t)row * IN_FEAT + threadIdx.x * 8) = z;
}

// stage one 32K x 128N bf16 B-tile into LDS (each of 256 threads moves 32B)
__device__ __forceinline__ void stage_B(const __bf16* __restrict__ Wexp, int nBase, int k,
                                        __bf16* lbase, int t) {
    const int row  = t >> 1;          // 0..127  (N index within tile)
    const int half = t & 1;           // which 32B half of the 64B row
    const __bf16* g = Wexp + (size_t)(nBase + row) * IN_FEAT + k + half * 16;
    __bf16*       l = lbase + row * LDS_ROW + half * 16;
#if HAVE_ASYNC
    __builtin_amdgcn_global_load_async_to_lds_b128(
        (AS1 v4i_t*)g, (AS3 v4i_t*)l, 0, 0);
    __builtin_amdgcn_global_load_async_to_lds_b128(
        (AS1 v4i_t*)(g + 8), (AS3 v4i_t*)(l + 8), 0, 0);
#else
    *(bf16x8*)l       = *(const bf16x8*)g;
    *(bf16x8*)(l + 8) = *(const bf16x8*)(g + 8);
#endif
}

// Grouped GEMM: 256 threads = 8 waves; block tile M=64 x N=128.
// B double-buffered through LDS (async copy overlapped with WMMA); A direct from global.
// Wave (wm,wn) computes rows [wm*16,+16) x cols [wn*64,+64) = 4 x v_wmma_f32_16x16x32_bf16 per K-step.
__global__ void __launch_bounds__(256)
k_gemm(const __bf16* __restrict__ Abf, const __bf16* __restrict__ Wbf,
       const int* __restrict__ perm, const int* __restrict__ segStart,
       float* __restrict__ out, float* __restrict__ dump) {
    __shared__ __attribute__((aligned(32))) __bf16 sB[2][BN * LDS_ROW];  // 2 x 10KB

    const int rowBase = blockIdx.y * 64;
    if (rowBase >= segStart[NUM_EXPERT]) return;        // uniform early-out past padded rows
    int e = 0;
#pragma unroll
    for (int i = 1; i < NUM_EXPERT; ++i) if (rowBase >= segStart[i]) e = i;

    const int nBase = blockIdx.x * BN;
    const int t     = threadIdx.x;
    const int wave  = t >> 5;
    const int lane  = t & 31;
    const int wm    = wave & 3;        // M subtile
    const int wn    = wave >> 2;       // N half
    const int lrow  = lane & 15;       // M index (A) / N index (B,D)
    const int lhalf = lane >> 4;       // K-chunk select

    const int row0 = rowBase + wm * 16;
    // A layout (16-bit, 16x32): lanes 0-15 K chunks {0..7,16..23}, lanes 16-31 {8..15,24..31}
    const __bf16* Arow = Abf + (size_t)(row0 + lrow) * IN_FEAT + lhalf * 8;
    const __bf16* Wexp = Wbf + (size_t)e * OUT_FEAT * IN_FEAT;

    f32x8 acc[4];
#pragma unroll
    for (int s = 0; s < 4; ++s)
#pragma unroll
        for (int v = 0; v < 8; ++v) acc[s][v] = 0.0f;

    stage_B(Wexp, nBase, 0, &sB[0][0], t);              // prologue: fill buffer 0

    const int NT = IN_FEAT / K_STEP;
#pragma unroll 1
    for (int kt = 0; kt < NT; ++kt) {
        const int k   = kt * K_STEP;
        const int buf = kt & 1;
        WAIT_ASYNC();                                   // this wave's async copies landed
        __syncthreads();                                // whole tile visible; prev reads done
        if (kt + 1 < NT)
            stage_B(Wexp, nBase, k + K_STEP, &sB[buf ^ 1][0], t);  // overlap next tile

        bf16x8 a0 = *(const bf16x8*)(Arow + k);         // K = k + h*8 .. +7
        bf16x8 a1 = *(const bf16x8*)(Arow + k + 16);    // K = k + h*8 + 16 .. +23
        bf16x16 a = __builtin_shufflevector(a0, a1,
                        0,1,2,3,4,5,6,7,8,9,10,11,12,13,14,15);

        const __bf16* sbuf = &sB[buf][(wn * 64) * LDS_ROW];
#pragma unroll
        for (int s = 0; s < 4; ++s) {
            // B layout (32x16): lane holds column lrow; lanes 0-15 K {0..7,16..23}, lanes 16-31 {8..15,24..31}
            const __bf16* bp = sbuf + (s * 16 + lrow) * LDS_ROW + lhalf * 8;
            bf16x8 b0 = *(const bf16x8*)bp;
            bf16x8 b1 = *(const bf16x8*)(bp + 16);
            bf16x16 b = __builtin_shufflevector(b0, b1,
                            0,1,2,3,4,5,6,7,8,9,10,11,12,13,14,15);
            acc[s] = __builtin_amdgcn_wmma_f32_16x16x32_bf16(
                         false, a, false, b, (short)0, acc[s], false, false);
        }
    }

    // D layout: VGPR v -> row v + 8*lhalf, col lrow. Branch-free scatter:
    // pad rows (perm == -1) store to a dump line via address select (v_cndmask, no saveexec).
    int tok[8];
#pragma unroll
    for (int v = 0; v < 8; ++v) tok[v] = perm[row0 + v + 8 * lhalf];
#pragma unroll
    for (int s = 0; s < 4; ++s) {
        const int col = nBase + wn * 64 + s * 16 + lrow;
#pragma unroll
        for (int v = 0; v < 8; ++v) {
            float* p = (tok[v] >= 0) ? (out + (size_t)tok[v] * OUT_FEAT + col)
                                     : (dump + lane);
            *p = acc[s][v];
        }
    }
}

extern "C" void kernel_launch(void* const* d_in, const int* in_sizes, int n_in,
                              void* d_out, int out_size, void* d_ws, size_t ws_size,
                              hipStream_t stream) {
    const float* inp    = (const float*)d_in[0];
    const int*   gate   = (const int*)d_in[1];
    const float* weight = (const float*)d_in[2];
    float*       out    = (float*)d_out;

    char* ws = (char*)d_ws;
    int* meta     = (int*)ws;
    int* counts   = meta + 0;
    int* fill     = meta + 8;
    int* segStart = meta + 16;
    int* perm     = meta + 32;
    __bf16* Abf  = (__bf16*)(ws + OFF_ABF);
    __bf16* Wbf  = (__bf16*)(ws + OFF_WBF);
    float*  dump = (float*)(ws + OFF_DUMP);

    k_init   <<<1, 32, 0, stream>>>(meta);
    k_count  <<<N_TOKENS / 256, 256, 0, stream>>>(gate, counts);
    k_scan   <<<1, 1, 0, stream>>>(counts, segStart);
    k_wconv  <<<(NUM_EXPERT * OUT_FEAT * IN_FEAT / 8) / 256, 256, 0, stream>>>(weight, Wbf);
    k_gather <<<N_TOKENS, 128, 0, stream>>>(inp, gate, segStart, fill, perm, Abf);
    k_padzero<<<MAX_ROWS, 128, 0, stream>>>(segStart, counts, perm, Abf);
    k_gemm   <<<dim3(OUT_FEAT / BN, MAX_MTILES), 256, 0, stream>>>(Abf, Wbf, perm, segStart, out, dump);
}
